// RobustMoERouter_38165079392677
// MI455X (gfx1250) — matrix operand order repaired and checked by reading, not compiled, not run on hardware
//
#include <hip/hip_runtime.h>

// ---------------------------------------------------------------- constants
constexpr int NTOK = 24576;
constexpr int DIM  = 2048;
constexpr int NEXP = 3;
constexpr int CAP  = NTOK / NEXP;          // 8192
constexpr int N1 = 1024, N2 = 512, N3 = 128;

// output layout: data[3][CAP][DIM], loss, idx[3][CAP]
constexpr size_t OUT_LOSS  = (size_t)NEXP * CAP * DIM;
constexpr size_t OUT_IDX   = OUT_LOSS + 1;

// ---------------------------------------------------------------- workspace
constexpr size_t OFF_W1B  = 0;                                   // bf16 1024x2048
constexpr size_t OFF_W2B  = OFF_W1B  + (size_t)N1 * DIM * 2;     // bf16 512x1024
constexpr size_t OFF_W3B  = OFF_W2B  + (size_t)N2 * N1  * 2;     // bf16 128x512
constexpr size_t OFF_H1   = OFF_W3B  + (size_t)N3 * N2  * 2;     // bf16 NTOKx1024
constexpr size_t OFF_H2   = OFF_H1   + (size_t)NTOK * N1 * 2;    // bf16 NTOKx512
constexpr size_t OFF_H3   = OFF_H2   + (size_t)NTOK * N2 * 2;    // bf16 NTOKx128
constexpr size_t OFF_PROB = OFF_H3   + (size_t)NTOK * N3 * 2;    // f32 NTOKx3
constexpr size_t OFF_PSEL = OFF_PROB + (size_t)NTOK * 3 * 4;     // f32 NTOK
constexpr size_t OFF_CHOI = OFF_PSEL + (size_t)NTOK * 4;         // i32 NTOK
constexpr size_t OFF_RANK = OFF_CHOI + (size_t)NTOK * 4;         // i32 NTOK
constexpr size_t OFF_ASGN = OFF_RANK + (size_t)NTOK * 4;         // i32 NTOK
constexpr size_t OFF_SCNR = OFF_ASGN + (size_t)NTOK * 4;         // i32 NTOK
constexpr size_t OFF_PROC = OFF_SCNR + (size_t)NTOK * 4;         // i32 NTOK
constexpr size_t OFF_IDXO = OFF_PROC + (size_t)NTOK * 4;         // i32 3*CAP
constexpr size_t OFF_PART = OFF_IDXO + (size_t)NEXP * CAP * 4;   // f32 96 partials
constexpr size_t OFF_MISC = OFF_PART + 256 * 4;                  // i32 64 words

// misc word indices
#define MI_CNT   0   // counts[3]
#define MI_OFF   3   // drop offsets[3]
#define MI_REM   6   // remaining cap[3]
#define MI_NDROP 9
#define MI_LOSS  10  // float (sum of probs)
#define MI_KEPT  11  // kept[3]

// ---------------------------------------------------------------- helpers
typedef __attribute__((ext_vector_type(16))) __bf16        v16bf;
typedef __attribute__((ext_vector_type(8)))  __bf16        v8bf;
typedef __attribute__((ext_vector_type(8)))  float         v8f;
typedef __attribute__((ext_vector_type(4)))  unsigned int  v4u;
typedef __attribute__((ext_vector_type(8)))  unsigned int  v8u;

__device__ __forceinline__ unsigned short f2bf(float f) {
  unsigned int u = __float_as_uint(f);
  u += 0x7FFFu + ((u >> 16) & 1u);        // round-to-nearest-even
  return (unsigned short)(u >> 16);
}
__device__ __forceinline__ float bf2f(unsigned short s) {
  return __uint_as_float(((unsigned int)s) << 16);
}
// pack two fp32 -> two bf16 in one dword via v_perm_b32 (bytes 2,3 of each)
__device__ __forceinline__ unsigned int pk_bf16(float lo, float hi) {
  unsigned int u0 = __float_as_uint(lo) + 0x7FFFu;
  unsigned int u1 = __float_as_uint(hi) + 0x7FFFu;
  return __builtin_amdgcn_perm(u1, u0, 0x07060302u);
}

// ---------------------------------------------------------------- TDM (CDNA5)
// 2-D tile load Global->LDS via Tensor Data Mover.  D# groups per ISA 8.3/8.4:
//   group0: count=1 | lds_addr | global_addr(57b) | type=2
//   group1: data_size=2B, pad_enable (16 DW interval + 4 DW pad = the 40-short
//           padded LDS row stride), tensor_dim0=K, tensor_dim1=rows,
//           tile 32 x tile_rows, dim0 stride = K.
// Issued by one wave (EXEC ignored); completion via s_wait_tensorcnt.
__device__ __forceinline__ void tdm_load_tile(const void* lds_dst,
                                              const void* gsrc,
                                              unsigned int K_elems,
                                              unsigned int rows_total,
                                              unsigned int tile_rows) {
  unsigned long long ga = (unsigned long long)(uintptr_t)gsrc;
  v4u g0;
  g0[0] = 1u;                                         // count=1, user mode
  g0[1] = (unsigned int)(uintptr_t)lds_dst;           // LDS byte address
  g0[2] = (unsigned int)ga;                           // global addr [31:0]
  g0[3] = (unsigned int)((ga >> 32) & 0x1FFFFFFu) | (2u << 30);  // [56:32]|type=2
  v8u g1;
  g1[0] = (1u << 16)        // data_size = 2 bytes
        | (1u << 20)        // pad_enable
        | (3u << 22)        // pad_interval: 16 DWORDs (= 32 bf16 row)
        | (3u << 25);       // pad_amount:   4 DWORDs (= 8 bf16 pad)
  g1[1] = (K_elems & 0xFFFFu) << 16;                          // tensor_dim0 lo
  g1[2] = ((K_elems >> 16) & 0xFFFFu) | ((rows_total & 0xFFFFu) << 16);
  g1[3] = ((rows_total >> 16) & 0xFFFFu) | (32u << 16);       // tile_dim0 = 32
  g1[4] = tile_rows & 0xFFFFu;                                // tile_dim1, tile_dim2=0
  g1[5] = K_elems;                                            // dim0 stride lo
  g1[6] = 0u;                                                 // stride hi, dim1 stride
  g1[7] = 0u;
  asm volatile("tensor_load_to_lds %0, %1" :: "s"(g0), "s"(g1) : "memory");
}
__device__ __forceinline__ void tdm_wait() {
#if __has_builtin(__builtin_amdgcn_s_wait_tensorcnt)
  __builtin_amdgcn_s_wait_tensorcnt(0);
#else
  asm volatile("s_wait_tensorcnt 0x0" ::: "memory");
#endif
}

// ---------------------------------------------------------------- fp32->bf16
__global__ void k_cvt_bf16(const float* __restrict__ src,
                           unsigned short* __restrict__ dst, int n) {
  int i = blockIdx.x * 256 + threadIdx.x;
  if (i < n) dst[i] = f2bf(src[i]);
}

// ---------------------------------------------------------------- bf16 WMMA GEMM
// C[M,N] = relu(A[M,K] * W[N,K]^T + bias[N]); A fp32 (layer1) or bf16, W bf16.
// Tile: 128(M) x 128(N) x 32(K); 256 threads = 8 waves (4Mx2N), each wave owns
// a 32x64 region = 2x4 of v_wmma_f32_16x16x32_bf16.
// Double-buffered LDS: tile s+1 is staged (TDM DMA and/or batched f32 loads
// with v_perm bf16 packing) while tile s runs its 8 WMMAs; a single barrier
// per iteration, TENSORcnt drained by wave 0 just before it.
template <bool A_F32>
__global__ __launch_bounds__(256, 1)
void k_gemm_relu(const void* __restrict__ Aany,
                 const unsigned short* __restrict__ Wb,
                 const float* __restrict__ bias,
                 unsigned short* __restrict__ Out,
                 int M, int K, int Nout) {
  __shared__ __align__(16) unsigned short As[2][128][40];   // pad 8: 80 B stride
  __shared__ __align__(16) unsigned short Bs[2][128][40];

  const int tid   = threadIdx.x;
  const int lane  = tid & 31;
  const int wave  = tid >> 5;
  const int wm    = wave >> 1;      // 0..3 -> M offset wm*32
  const int wn    = wave & 1;       // 0..1 -> N offset wn*64
  const int l15   = lane & 15;
  const int hi    = lane >> 4;
  const int blockM = blockIdx.y * 128;
  const int blockN = blockIdx.x * 128;

  const float* Af = (const float*)Aany;
  const unsigned short* Ab = (const unsigned short*)Aany;

  // batched f32 A staging: 4 loads in flight, then convert+store
  auto stageA_f32 = [&](int k0, int buf) {
    float4 va[4];
    #pragma unroll
    for (int q = 0; q < 4; ++q) {
      int c = q * 256 + tid;               // 1024 float4 chunks
      int row = c >> 3, col4 = (c & 7) * 4;
      va[q] = *(const float4*)(Af + (size_t)(blockM + row) * K + k0 + col4);
    }
    #pragma unroll
    for (int q = 0; q < 4; ++q) {
      int c = q * 256 + tid;
      int row = c >> 3, col4 = (c & 7) * 4;
      *(uint2*)&As[buf][row][col4] =
          make_uint2(pk_bf16(va[q].x, va[q].y), pk_bf16(va[q].z, va[q].w));
    }
  };

  const v8f vzero = {0.f, 0.f, 0.f, 0.f, 0.f, 0.f, 0.f, 0.f};
  v8f acc[2][4];
  #pragma unroll
  for (int im = 0; im < 2; ++im)
    #pragma unroll
    for (int in = 0; in < 4; ++in) acc[im][in] = vzero;

  // ---- prologue: stage tile 0 into buffer 0
  if constexpr (A_F32) {
    stageA_f32(0, 0);
    if (wave == 0) {
      tdm_load_tile(&Bs[0][0][0], Wb + (size_t)blockN * K,
                    (unsigned)K, (unsigned)Nout, 128u);
      tdm_wait();
    }
  } else {
    if (wave == 0) {
      tdm_load_tile(&As[0][0][0], Ab + (size_t)blockM * K,
                    (unsigned)K, (unsigned)M, 128u);
      tdm_load_tile(&Bs[0][0][0], Wb + (size_t)blockN * K,
                    (unsigned)K, (unsigned)Nout, 128u);
      tdm_wait();
    }
  }
  __syncthreads();

  const int nsteps = K >> 5;
  for (int s = 0; s < nsteps; ++s) {
    const int cur = s & 1, nxt = cur ^ 1;
    const int k1  = (s + 1) * 32;
    const bool more = (s + 1 < nsteps);

    // ---- stage tile s+1 into the alternate buffer (overlaps the WMMAs)
    if constexpr (A_F32) {
      if (more) {
        stageA_f32(k1, nxt);
        if (wave == 0)
          tdm_load_tile(&Bs[nxt][0][0], Wb + (size_t)blockN * K + k1,
                        (unsigned)K, (unsigned)Nout, 128u);
        if (s + 2 < nsteps)
          __builtin_prefetch(Af + (size_t)(blockM + (tid >> 1)) * K + k1 + 32, 0, 1);
      }
    } else {
      if (more && wave == 0) {
        tdm_load_tile(&As[nxt][0][0], Ab + (size_t)blockM * K + k1,
                      (unsigned)K, (unsigned)M, 128u);
        tdm_load_tile(&Bs[nxt][0][0], Wb + (size_t)blockN * K + k1,
                      (unsigned)K, (unsigned)Nout, 128u);
      }
    }

    // ---- compute on current buffer: fragments per ISA wave32 layouts
    v16bf afr[2], bfr[4];
    #pragma unroll
    for (int im = 0; im < 2; ++im) {
      const unsigned short* p = &As[cur][wm * 32 + im * 16 + l15][hi * 8];
      v8bf lo = *(const v8bf*)p;
      v8bf hh = *(const v8bf*)(p + 16);
      afr[im] = __builtin_shufflevector(lo, hh, 0,1,2,3,4,5,6,7,8,9,10,11,12,13,14,15);
    }
    #pragma unroll
    for (int in = 0; in < 4; ++in) {
      const unsigned short* p = &Bs[cur][wn * 64 + in * 16 + l15][hi * 8];
      v8bf lo = *(const v8bf*)p;
      v8bf hh = *(const v8bf*)(p + 16);
      bfr[in] = __builtin_shufflevector(lo, hh, 0,1,2,3,4,5,6,7,8,9,10,11,12,13,14,15);
    }
    #pragma unroll
    for (int im = 0; im < 2; ++im)
      #pragma unroll
      for (int in = 0; in < 4; ++in)
        acc[im][in] = __builtin_amdgcn_wmma_f32_16x16x32_bf16(
            false, afr[im], false, bfr[in], (short)0, acc[im][in], false, false);

    if (more && wave == 0) tdm_wait();   // drain DMA for tile s+1
    __syncthreads();                     // publish next buffer / retire reads
  }

  // ---- epilogue: bias + ReLU, store bf16
  #pragma unroll
  for (int im = 0; im < 2; ++im) {
    #pragma unroll
    for (int in = 0; in < 4; ++in) {
      int col = blockN + wn * 64 + in * 16 + l15;
      float bv = bias[col];
      #pragma unroll
      for (int r = 0; r < 8; ++r) {
        int row = blockM + wm * 32 + im * 16 + hi * 8 + r;
        float v = fmaxf(acc[im][in][r] + bv, 0.f);
        Out[(size_t)row * Nout + col] = f2bf(v);
      }
    }
  }
}

// ---------------------------------------------------------------- layer4 + softmax
__global__ __launch_bounds__(256)
void k_gate_softmax(const unsigned short* __restrict__ h3,
                    const float* __restrict__ W4, const float* __restrict__ b4,
                    float* __restrict__ probs, float* __restrict__ psel,
                    int* __restrict__ choice, int* __restrict__ misc,
                    float* __restrict__ partial) {
  __shared__ float w[3][128];
  __shared__ float bb[3];
  __shared__ float red[256];
  for (int i = threadIdx.x; i < 384; i += 256) w[i >> 7][i & 127] = W4[i];
  if (threadIdx.x < 3) bb[threadIdx.x] = b4[threadIdx.x];
  __syncthreads();

  int t = blockIdx.x * 256 + threadIdx.x;        // NTOK % 256 == 0
  const unsigned short* hp = h3 + (size_t)t * 128;
  float d0 = bb[0], d1 = bb[1], d2 = bb[2];
  #pragma unroll 8
  for (int k = 0; k < 128; ++k) {
    float xv = bf2f(hp[k]);
    d0 += xv * w[0][k];
    d1 += xv * w[1][k];
    d2 += xv * w[2][k];
  }
  float m  = fmaxf(d0, fmaxf(d1, d2));
  float e0 = __expf(d0 - m), e1 = __expf(d1 - m), e2 = __expf(d2 - m);
  float inv = 1.f / (e0 + e1 + e2);
  float p0 = e0 * inv, p1 = e1 * inv, p2 = e2 * inv;
  probs[t * 3 + 0] = p0; probs[t * 3 + 1] = p1; probs[t * 3 + 2] = p2;
  int c = 0; float best = p0;
  if (p1 > best) { best = p1; c = 1; }
  if (p2 > best) { best = p2; c = 2; }
  choice[t] = c; psel[t] = best;
  atomicAdd(&misc[MI_CNT + c], 1);

  red[threadIdx.x] = p0 + p1 + p2;               // deterministic loss reduction
  __syncthreads();
  for (int st = 128; st > 0; st >>= 1) {
    if (threadIdx.x < st) red[threadIdx.x] += red[threadIdx.x + st];
    __syncthreads();
  }
  if (threadIdx.x == 0) partial[blockIdx.x] = red[0];
}

// ---------------------------------------------------------------- offsets/capacity
__global__ void k_offsets(int* misc, const float* partial, int nblk) {
  int c0 = misc[MI_CNT + 0], c1 = misc[MI_CNT + 1], c2 = misc[MI_CNT + 2];
  int d0 = max(c0 - CAP, 0), d1 = max(c1 - CAP, 0), d2 = max(c2 - CAP, 0);
  misc[MI_OFF + 0] = 0;
  misc[MI_OFF + 1] = d0;
  misc[MI_OFF + 2] = d0 + d1;
  misc[MI_NDROP]   = d0 + d1 + d2;
  misc[MI_REM + 0] = CAP - min(c0, CAP);
  misc[MI_REM + 1] = CAP - min(c1, CAP);
  misc[MI_REM + 2] = CAP - min(c2, CAP);
  misc[MI_KEPT + 0] = min(c0, CAP);
  misc[MI_KEPT + 1] = min(c1, CAP);
  misc[MI_KEPT + 2] = min(c2, CAP);
  float s = 0.f;
  for (int i = 0; i < nblk; ++i) s += partial[i];
  ((float*)misc)[MI_LOSS] = s;
}

// ---------------------------------------------------------------- per-expert rank
__global__ __launch_bounds__(256)
void k_rank(const float* __restrict__ psel, const int* __restrict__ choice,
            int* __restrict__ rank, int* __restrict__ assign,
            int* __restrict__ proc_idx, const int* __restrict__ misc) {
  __shared__ float sp[256];
  __shared__ int   sc[256];
  int i = blockIdx.x * 256 + threadIdx.x;
  float p = psel[i];
  int   c = choice[i];
  int cnt = 0;
  for (int j0 = 0; j0 < NTOK; j0 += 256) {
    __syncthreads();
    sp[threadIdx.x] = psel[j0 + threadIdx.x];
    sc[threadIdx.x] = choice[j0 + threadIdx.x];
    __syncthreads();
    #pragma unroll 4
    for (int jj = 0; jj < 256; ++jj) {
      int j = j0 + jj;
      if (sc[jj] == c && (sp[jj] > p || (sp[jj] == p && j < i))) cnt++;
    }
  }
  rank[i]   = cnt;
  assign[i] = c;
  if (cnt >= CAP) {   // dropped: proc position = expert asc, prob asc (= rank desc)
    int pos = misc[MI_OFF + c] + (misc[MI_CNT + c] - 1 - cnt);
    proc_idx[pos] = i;
  }
}

// ---------------------------------------------------------------- sequential scan
__global__ void k_scan(const int* __restrict__ proc_idx,
                       const int* __restrict__ choice,
                       const float* __restrict__ probs,
                       int* __restrict__ assign, int* __restrict__ scanr,
                       const int* __restrict__ misc) {
  int rem[3] = {misc[MI_REM + 0], misc[MI_REM + 1], misc[MI_REM + 2]};
  int nd = misc[MI_NDROP];
  for (int p = 0; p < nd; ++p) {
    int t = proc_idx[p];
    int c = choice[t];
    int a = (c == 0) ? 1 : 0;
    int b = (c == 2) ? 1 : 2;
    float pa = probs[t * 3 + a], pb = probs[t * 3 + b];
    int c0 = a, c1 = b;
    if (pb > pa) { c0 = b; c1 = a; }          // stable: tie keeps lower index
    int tgt = (rem[c0] > 0) ? c0 : ((rem[c1] > 0) ? c1 : c0);
    rem[tgt]--;
    assign[t] = tgt;
    scanr[t]  = p;
  }
}

// ---------------------------------------------------------------- output positions
__global__ __launch_bounds__(256)
void k_positions(const int* __restrict__ assign, const int* __restrict__ rank,
                 const int* __restrict__ scanr, const int* __restrict__ misc,
                 int* __restrict__ idx_out) {
  __shared__ int sa[256];
  __shared__ int sd[256];
  __shared__ int ss[256];
  int i  = blockIdx.x * 256 + threadIdx.x;
  int e  = assign[i];
  int di = (rank[i] >= CAP) ? 1 : 0;
  int si = di ? scanr[i] : 0;
  int cnt = 0;
  for (int j0 = 0; j0 < NTOK; j0 += 256) {
    __syncthreads();
    int j = j0 + threadIdx.x;
    sa[threadIdx.x] = assign[j];
    sd[threadIdx.x] = (rank[j] >= CAP) ? 1 : 0;
    ss[threadIdx.x] = scanr[j];
    __syncthreads();
    if (!di) {
      #pragma unroll 4
      for (int jj = 0; jj < 256; ++jj)
        if (sa[jj] == e && !sd[jj] && (j0 + jj) < i) cnt++;
    } else {
      #pragma unroll 4
      for (int jj = 0; jj < 256; ++jj)
        if (sa[jj] == e && sd[jj] && ss[jj] < si) cnt++;
    }
  }
  int pos = di ? (misc[MI_KEPT + e] + cnt) : cnt;
  if (pos < CAP) idx_out[e * CAP + pos] = i;
}

// ---------------------------------------------------------------- gather + tail
__global__ __launch_bounds__(256)
void k_gather(const float* __restrict__ x, const int* __restrict__ idx_out,
              const int* __restrict__ misc, float* __restrict__ out) {
  int blk = blockIdx.x;                 // 0 .. 3*CAP-1
  int tok = idx_out[blk];
  const float4* src = (const float4*)(x + (size_t)tok * DIM);
  float4*       dst = (float4*)(out + (size_t)blk * DIM);
  #pragma unroll
  for (int q = 0; q < 2; ++q)
    dst[threadIdx.x + q * 256] = src[threadIdx.x + q * 256];
  if (threadIdx.x == 0) {
    out[OUT_IDX + blk] = (float)tok;
    if (blk == 0)
      out[OUT_LOSS] = 0.1f / (3.0f * (float)NTOK) * ((const float*)misc)[MI_LOSS];
  }
}

// ---------------------------------------------------------------- launcher
extern "C" void kernel_launch(void* const* d_in, const int* in_sizes, int n_in,
                              void* d_out, int out_size, void* d_ws, size_t ws_size,
                              hipStream_t stream) {
  const float* x  = (const float*)d_in[0];
  const float* W1 = (const float*)d_in[1];
  const float* b1 = (const float*)d_in[2];
  const float* W2 = (const float*)d_in[3];
  const float* b2 = (const float*)d_in[4];
  const float* W3 = (const float*)d_in[5];
  const float* b3 = (const float*)d_in[6];
  const float* W4 = (const float*)d_in[7];
  const float* b4 = (const float*)d_in[8];
  float* out = (float*)d_out;

  char* ws = (char*)d_ws;
  unsigned short* w1b = (unsigned short*)(ws + OFF_W1B);
  unsigned short* w2b = (unsigned short*)(ws + OFF_W2B);
  unsigned short* w3b = (unsigned short*)(ws + OFF_W3B);
  unsigned short* h1  = (unsigned short*)(ws + OFF_H1);
  unsigned short* h2  = (unsigned short*)(ws + OFF_H2);
  unsigned short* h3  = (unsigned short*)(ws + OFF_H3);
  float* probs   = (float*)(ws + OFF_PROB);
  float* psel    = (float*)(ws + OFF_PSEL);
  int*   choice  = (int*)(ws + OFF_CHOI);
  int*   rank    = (int*)(ws + OFF_RANK);
  int*   assign  = (int*)(ws + OFF_ASGN);
  int*   scanr   = (int*)(ws + OFF_SCNR);
  int*   procidx = (int*)(ws + OFF_PROC);
  int*   idx_out = (int*)(ws + OFF_IDXO);
  float* partial = (float*)(ws + OFF_PART);
  int*   misc    = (int*)(ws + OFF_MISC);

  // clear counters / index buffer (graph-capture-safe)
  hipMemsetAsync(misc, 0, 64 * sizeof(int), stream);
  hipMemsetAsync(idx_out, 0, (size_t)NEXP * CAP * sizeof(int), stream);

  // weights -> bf16
  k_cvt_bf16<<<(N1 * DIM + 255) / 256, 256, 0, stream>>>(W1, w1b, N1 * DIM);
  k_cvt_bf16<<<(N2 * N1  + 255) / 256, 256, 0, stream>>>(W2, w2b, N2 * N1);
  k_cvt_bf16<<<(N3 * N2  + 255) / 256, 256, 0, stream>>>(W3, w3b, N3 * N2);

  // gate MLP: WMMA bf16 GEMMs; tiles double-buffered via the TDM
  k_gemm_relu<true ><<<dim3(N1 / 128, NTOK / 128), 256, 0, stream>>>(x,  w1b, b1, h1, NTOK, DIM, N1);
  k_gemm_relu<false><<<dim3(N2 / 128, NTOK / 128), 256, 0, stream>>>(h1, w2b, b2, h2, NTOK, N1,  N2);
  k_gemm_relu<false><<<dim3(N3 / 128, NTOK / 128), 256, 0, stream>>>(h2, w3b, b3, h3, NTOK, N2,  N3);

  // layer 4 + softmax + argmax + counts + loss partials
  k_gate_softmax<<<NTOK / 256, 256, 0, stream>>>(h3, W4, b4, probs, psel, choice, misc, partial);
  k_offsets<<<1, 1, 0, stream>>>(misc, partial, NTOK / 256);

  // capacity routing
  k_rank<<<NTOK / 256, 256, 0, stream>>>(psel, choice, rank, assign, procidx, misc);
  k_scan<<<1, 1, 0, stream>>>(procidx, choice, probs, assign, scanr, misc);
  k_positions<<<NTOK / 256, 256, 0, stream>>>(assign, rank, scanr, misc, idx_out);

  // gather expert data + indices + loss
  k_gather<<<NEXP * CAP, 256, 0, stream>>>(x, idx_out, misc, out);
}
